// CohereASRMultiHeadAttention_6133213299428
// MI455X (gfx1250) — matrix operand order repaired and checked by reading, not compiled, not run on hardware
//
#include <hip/hip_runtime.h>
#include <hip/hip_bf16.h>

// MHA: B=8, T=2048, FDIM=512, H=8, DK=64.  fp32 I/O, bf16 WMMA internally.
#define B_    8
#define T_    2048
#define FDIM_ 512
#define H_    8
#define DK_   64
#define INF_VAL 10000.0f

typedef __attribute__((ext_vector_type(16))) __bf16 v16bf;
typedef __attribute__((ext_vector_type(8)))  float  v8f;

// Fragment viewed as two 16-byte quads -> two ds_load_b128 per fragment.
union FragB16 { v16bf v; uint4 q[2]; };
union Pack8   { __bf16 h[8]; uint4 q; };

__device__ __forceinline__ v8f wmma_bf16(v16bf a, v16bf b, v8f c) {
    // D = A(16x32 bf16) * B(32x16 bf16) + C(16x16 f32)
    return __builtin_amdgcn_wmma_f32_16x16x32_bf16(
        false, a, false, b, (short)0, c, false, false);
}

__device__ __forceinline__ v8f vzero8() {
    v8f z;
#pragma unroll
    for (int i = 0; i < 8; ++i) z[i] = 0.0f;
    return z;
}

// ---------------------------------------------------------------------------
// Tensor Data Mover: 2-D tile load Global -> LDS (descriptor per ISA ch.8).
// This toolchain exposes the 6-arg builtin:
//   (u32x4 g0, i32x8 g1, i32x4 g2, i32x4 g3, i32x8 pad, i32 cpol)
// ---------------------------------------------------------------------------
#if defined(__has_builtin)
#if __has_builtin(__builtin_amdgcn_tensor_load_to_lds) && \
    __has_builtin(__builtin_amdgcn_s_wait_tensorcnt)
#define USE_TDM 1
#endif
#endif

#ifdef USE_TDM
typedef unsigned int u32x4 __attribute__((ext_vector_type(4)));
typedef int          i32x4 __attribute__((ext_vector_type(4)));
typedef int          i32x8 __attribute__((ext_vector_type(8)));

// data_size_log: 0=1B,1=2B.  Dims/strides in data_size units.
__device__ __forceinline__ void tdm_load_2d(unsigned lds_off, const void* gptr,
                                            unsigned data_size_log,
                                            unsigned tdim0, unsigned tdim1,
                                            unsigned tile0, unsigned tile1,
                                            unsigned stride0)
{
    unsigned long long ga = (unsigned long long)(size_t)gptr;
    u32x4 g0;
    g0[0] = 1u;                                              // count=1, user mode
    g0[1] = lds_off;                                         // D#.lds_addr
    g0[2] = (unsigned)ga;                                    // global_addr[31:0]
    g0[3] = (unsigned)((ga >> 32) & 0x1FFFFFFu) | (2u << 30); // addr[56:32] | type=2
    i32x8 g1;
    g1[0] = (int)(data_size_log << 16);                      // wg_mask=0, data_size
    g1[1] = (int)((tdim0 & 0xFFFFu) << 16);                  // tensor_dim0[15:0]
    g1[2] = (int)((tdim0 >> 16) | ((tdim1 & 0xFFFFu) << 16));// td0[31:16] | td1[15:0]
    g1[3] = (int)((tdim1 >> 16) | (tile0 << 16));            // td1[31:16] | tile_dim0
    g1[4] = (int)tile1;                                      // tile_dim1 | tile_dim2=0
    g1[5] = (int)stride0;                                    // tensor_dim0_stride lo
    g1[6] = 0;                                               // stride0 hi | stride1 lo
    g1[7] = 0;                                               // stride1 hi
    i32x4 z4 = {0, 0, 0, 0};                                 // groups 2/3 unused (2-D)
    i32x8 z8 = {0, 0, 0, 0, 0, 0, 0, 0};
    __builtin_amdgcn_tensor_load_to_lds(g0, g1, z4, z4, z8, 0);
}
#endif

// A-fragment (16x32, 16-bit): lane's 16 elements are one contiguous 16-run:
//   k in [hi*16, hi*16+16), m = lane&15 (same k order as B)
// B-fragment (32x16): one contiguous 16-run: k in [hi*16, hi*16+16), n = lane&15.

// ---------------------------------------------------------------------------
// Tiled GEMM:  Out = X @ W^T + bias.   X:[M,K] (Tin), W:[N,K] f32 row-major.
// Block tile 64x64, 256 threads = 8 waves; wave tile 16x32 (2 WMMA / k-step).
// ---------------------------------------------------------------------------
template <typename Tin, bool TO_BHTD>
__global__ __launch_bounds__(256)
void gemm_wt_bias(const Tin* __restrict__ X,
                  const float* __restrict__ W,
                  const float* __restrict__ bias,
                  void* __restrict__ Out,
                  int M, int N, int K)
{
    __shared__ __bf16 sA[64][32];  // [m][k]
    __shared__ __bf16 sB[64][32];  // [n][k]  (B[k][n] = sB[n][k])

    const int tid  = threadIdx.x;
    const int lane = tid & 31;
    const int wave = tid >> 5;
    const int hi   = lane >> 4;
    const int ln   = lane & 15;
    const int wm   = wave & 3;
    const int wn   = wave >> 2;
    const int m0   = blockIdx.x * 64;
    const int n0   = blockIdx.y * 64;

    const int sr = tid >> 2;           // staging row 0..63
    const int sc = (tid & 3) * 8;      // staging col {0,8,16,24}

    v8f acc[2] = { vzero8(), vzero8() };

    for (int k0 = 0; k0 < K; k0 += 32) {
        // Stage A tile (64x32) -> bf16, one b128 store per thread.
        {
            const Tin* src = X + (size_t)(m0 + sr) * K + (k0 + sc);
            if constexpr (sizeof(Tin) == 4) {
                float4 f0 = *(const float4*)(src);
                float4 f1 = *(const float4*)(src + 4);
                Pack8 p;
                p.h[0] = (__bf16)f0.x; p.h[1] = (__bf16)f0.y;
                p.h[2] = (__bf16)f0.z; p.h[3] = (__bf16)f0.w;
                p.h[4] = (__bf16)f1.x; p.h[5] = (__bf16)f1.y;
                p.h[6] = (__bf16)f1.z; p.h[7] = (__bf16)f1.w;
                *(uint4*)&sA[sr][sc] = p.q;
            } else {
                *(uint4*)&sA[sr][sc] = *(const uint4*)src;
            }
        }
        // Stage W tile (64n x 32k) -> bf16.
        {
            const float* src = W + (size_t)(n0 + sr) * K + (k0 + sc);
            float4 f0 = *(const float4*)(src);
            float4 f1 = *(const float4*)(src + 4);
            Pack8 p;
            p.h[0] = (__bf16)f0.x; p.h[1] = (__bf16)f0.y;
            p.h[2] = (__bf16)f0.z; p.h[3] = (__bf16)f0.w;
            p.h[4] = (__bf16)f1.x; p.h[5] = (__bf16)f1.y;
            p.h[6] = (__bf16)f1.z; p.h[7] = (__bf16)f1.w;
            *(uint4*)&sB[sr][sc] = p.q;
        }
        __syncthreads();

        FragB16 a, b0, b1;
        a.q[0]  = *(const uint4*)&sA[wm * 16 + ln][hi * 16];
        a.q[1]  = *(const uint4*)&sA[wm * 16 + ln][hi * 16 + 8];
        b0.q[0] = *(const uint4*)&sB[wn * 32 +      ln][hi * 16];
        b0.q[1] = *(const uint4*)&sB[wn * 32 +      ln][hi * 16 + 8];
        b1.q[0] = *(const uint4*)&sB[wn * 32 + 16 + ln][hi * 16];
        b1.q[1] = *(const uint4*)&sB[wn * 32 + 16 + ln][hi * 16 + 8];

        acc[0] = wmma_bf16(a.v, b0.v, acc[0]);
        acc[1] = wmma_bf16(a.v, b1.v, acc[1]);
        __syncthreads();
    }

    // Epilogue. C-frag: element r -> (m = r + 8*hi, n = lane&15).
#pragma unroll
    for (int f = 0; f < 2; ++f) {
#pragma unroll
        for (int r = 0; r < 8; ++r) {
            const int gm = m0 + wm * 16 + r + hi * 8;
            const int gn = n0 + wn * 32 + f * 16 + ln;
            const float v = acc[f][r] + bias[gn];
            if (TO_BHTD) {
                const int bb = gm / T_, tt = gm % T_;
                const int hh = gn / DK_, dd = gn % DK_;
                ((__bf16*)Out)[(((size_t)bb * H_ + hh) * T_ + tt) * DK_ + dd] = (__bf16)v;
            } else {
                ((float*)Out)[(size_t)gm * N + gn] = v;
            }
        }
    }
}

// ---------------------------------------------------------------------------
// Flash attention: 128 query rows/block (16/wave), 32-key tiles via LDS.
// K tile + mask slab staged by the Tensor Data Mover when available.
// ---------------------------------------------------------------------------
__global__ __launch_bounds__(256)
void flash_attn(const __bf16* __restrict__ Q,
                const __bf16* __restrict__ Kq,
                const __bf16* __restrict__ V,
                const unsigned char* __restrict__ mask,  // [B,T,T] bool
                __bf16* __restrict__ Xout)
{
    __shared__ __bf16 sK[32][DK_];          // keys x dims (4 KB)
    __shared__ __bf16 sVt[DK_][32];         // dims x keys (transposed, 4 KB)
    __shared__ unsigned char sM[128][32];   // mask slab (4 KB)
    __shared__ __bf16 sP[8][16][32];        // per-wave P tiles (8 KB)

    const int tid  = threadIdx.x;
    const int lane = tid & 31;
    const int wave = tid >> 5;
    const int hi   = lane >> 4;
    const int ln   = lane & 15;
    const int b    = blockIdx.z;
    const int h    = blockIdx.y;
    const int q0   = blockIdx.x * 128;
    const int qw   = q0 + wave * 16;

    const __bf16* Qb = Q  + ((size_t)b * H_ + h) * T_ * DK_;
    const __bf16* Kb = Kq + ((size_t)b * H_ + h) * T_ * DK_;
    const __bf16* Vb = V  + ((size_t)b * H_ + h) * T_ * DK_;
    const unsigned char* Mb = mask + (size_t)b * T_ * T_;

    // Preload Q A-fragments: 16 rows x DK=64 -> two K=32 chunks, b128 loads.
    FragB16 aq[2];
#pragma unroll
    for (int c = 0; c < 2; ++c) {
        const __bf16* qr = Qb + (size_t)(qw + ln) * DK_ + c * 32;
        aq[c].q[0] = *(const uint4*)(qr + hi * 16);
        aq[c].q[1] = *(const uint4*)(qr + hi * 16 + 8);
    }

    v8f o[4] = { vzero8(), vzero8(), vzero8(), vzero8() };
    float mrun[8], lrun[8];
#pragma unroll
    for (int r = 0; r < 8; ++r) { mrun[r] = -3.0e38f; lrun[r] = 0.0f; }

    const float scale = 0.125f;  // 1/sqrt(DK)

#ifdef USE_TDM
    const unsigned sK_off = (unsigned)(size_t)&sK[0][0];  // flat addr[31:0] = LDS offset
    const unsigned sM_off = (unsigned)(size_t)&sM[0][0];
#endif

    for (int j0 = 0; j0 < T_; j0 += 32) {
#ifdef USE_TDM
        // TDM stages K tile (32 keys x 64 bf16) and mask slab (128 x 32 B).
        if (wave == 0) {
            tdm_load_2d(sK_off, Kb + (size_t)j0 * DK_, /*2B*/1,
                        DK_, T_, DK_, 32, DK_);
        } else if (wave == 1) {
            tdm_load_2d(sM_off, Mb + (size_t)q0 * T_ + j0, /*1B*/0,
                        T_, T_, 32, 128, T_);
        }
#else
        // Manual K staging: b128 copies.
        {
            const int r = tid >> 3, c = (tid & 7) * 8;
            *(uint4*)&sK[r][c] = *(const uint4*)&Kb[(size_t)(j0 + r) * DK_ + c];
        }
        // Manual mask staging: 16 B/thread.
        {
            const int r = tid >> 1, c = (tid & 1) * 16;
            *(uint4*)&sM[r][c] = *(const uint4*)&Mb[(size_t)(q0 + r) * T_ + (j0 + c)];
        }
#endif
        // V staged transposed (TDM cannot transpose): load b128, scatter b16.
        {
            const int r = tid >> 3, c = (tid & 7) * 8;
            Pack8 p;
            p.q = *(const uint4*)&Vb[(size_t)(j0 + r) * DK_ + c];
#pragma unroll
            for (int i = 0; i < 8; ++i) sVt[c + i][r] = p.h[i];
        }
        if (j0 + 32 < T_) {  // prefetch next K/V rows (global_prefetch_b8)
            const size_t nxt = (size_t)(j0 + 32 + (tid >> 3)) * DK_;
            __builtin_prefetch(&Kb[nxt], 0, 1);
            __builtin_prefetch(&Vb[nxt], 0, 1);
        }
#ifdef USE_TDM
        if (wave < 2) __builtin_amdgcn_s_wait_tensorcnt(0);
#endif
        __syncthreads();

        // S = Q K^T : 2 key fragments x 2 WMMA (DK=64).
        v8f s[2];
#pragma unroll
        for (int jf = 0; jf < 2; ++jf) {
            v8f accS = vzero8();
#pragma unroll
            for (int c = 0; c < 2; ++c) {
                FragB16 bk;
                const __bf16* kr = &sK[jf * 16 + ln][c * 32 + hi * 16];
                bk.q[0] = *(const uint4*)(kr);
                bk.q[1] = *(const uint4*)(kr + 8);
                accS = wmma_bf16(aq[c].v, bk.v, accS);
            }
            s[jf] = accS;
        }

        // Online softmax; fragment row r lives in one element across 16 lanes.
#pragma unroll
        for (int r = 0; r < 8; ++r) {
            const int qrow = wave * 16 + r + hi * 8;
            float s0 = s[0][r] * scale;
            float s1 = s[1][r] * scale;
            const bool msk0 = sM[qrow][ln]      != 0;
            const bool msk1 = sM[qrow][16 + ln] != 0;
            if (msk0) s0 = -INF_VAL;
            if (msk1) s1 = -INF_VAL;

            float tmax = fmaxf(s0, s1);
#pragma unroll
            for (int off = 1; off < 16; off <<= 1)
                tmax = fmaxf(tmax, __shfl_xor(tmax, off, 32));

            const float mnew  = fmaxf(mrun[r], tmax);
            const float alpha = __expf(mrun[r] - mnew);
            float p0 = msk0 ? 0.0f : __expf(s0 - mnew);   // reference zeroes masked attn
            float p1 = msk1 ? 0.0f : __expf(s1 - mnew);
            float psum = p0 + p1;
#pragma unroll
            for (int off = 1; off < 16; off <<= 1)
                psum += __shfl_xor(psum, off, 32);

            lrun[r] = lrun[r] * alpha + psum;
            mrun[r] = mnew;
#pragma unroll
            for (int f = 0; f < 4; ++f) o[f][r] *= alpha;

            sP[wave][r + hi * 8][ln]      = (__bf16)p0;
            sP[wave][r + hi * 8][16 + ln] = (__bf16)p1;
        }

        // P as A-fragment (same-wave LDS ops are in-order).
        FragB16 ap;
        ap.q[0] = *(const uint4*)&sP[wave][ln][hi * 16];
        ap.q[1] = *(const uint4*)&sP[wave][ln][hi * 16 + 8];

        // O += P V : V^T rows give contiguous B-fragment runs.
#pragma unroll
        for (int f = 0; f < 4; ++f) {
            FragB16 bv;
            const __bf16* vr = &sVt[f * 16 + ln][hi * 16];
            bv.q[0] = *(const uint4*)(vr);
            bv.q[1] = *(const uint4*)(vr + 8);
            o[f] = wmma_bf16(ap.v, bv.v, o[f]);
        }
        __syncthreads();
    }

    // Normalize and write bf16 output at [b, t, h*DK + d].
#pragma unroll
    for (int r = 0; r < 8; ++r) {
        const float inv = 1.0f / fmaxf(lrun[r], 1e-20f);
        const int t = qw + r + hi * 8;
#pragma unroll
        for (int f = 0; f < 4; ++f) {
            const float v = o[f][r] * inv;
            Xout[((size_t)b * T_ + t) * FDIM_ + h * DK_ + f * 16 + ln] = (__bf16)v;
        }
    }
}

// ---------------------------------------------------------------------------
extern "C" void kernel_launch(void* const* d_in, const int* in_sizes, int n_in,
                              void* d_out, int out_size, void* d_ws, size_t ws_size,
                              hipStream_t stream) {
    (void)in_sizes; (void)n_in; (void)out_size; (void)ws_size;
    const float* query = (const float*)d_in[0];
    const float* key_i = (const float*)d_in[1];
    const float* value = (const float*)d_in[2];
    const unsigned char* mask = (const unsigned char*)d_in[3];  // jnp bool -> 1 byte
    const float* Wq = (const float*)d_in[4];  const float* bq = (const float*)d_in[5];
    const float* Wk = (const float*)d_in[6];  const float* bk = (const float*)d_in[7];
    const float* Wv = (const float*)d_in[8];  const float* bv = (const float*)d_in[9];
    const float* Wo = (const float*)d_in[10]; const float* bo = (const float*)d_in[11];

    const size_t per = (size_t)B_ * H_ * T_ * DK_;
    __bf16* q_ws = (__bf16*)d_ws;
    __bf16* k_ws = q_ws + per;
    __bf16* v_ws = k_ws + per;
    __bf16* x_ws = v_ws + per;

    const int M = B_ * T_;     // 16384
    const int N = FDIM_;       // 512
    const int K = FDIM_;       // 512
    dim3 gblk(256);
    dim3 ggrid(M / 64, N / 64);

    gemm_wt_bias<float, true><<<ggrid, gblk, 0, stream>>>(query, Wq, bq, (void*)q_ws, M, N, K);
    gemm_wt_bias<float, true><<<ggrid, gblk, 0, stream>>>(key_i, Wk, bk, (void*)k_ws, M, N, K);
    gemm_wt_bias<float, true><<<ggrid, gblk, 0, stream>>>(value, Wv, bv, (void*)v_ws, M, N, K);

    dim3 agrid(T_ / 128, H_, B_);
    flash_attn<<<agrid, gblk, 0, stream>>>(q_ws, k_ws, v_ws, mask, x_ws);

    gemm_wt_bias<__bf16, false><<<ggrid, gblk, 0, stream>>>(x_ws, Wo, bo, d_out, M, N, K);
}
